// MultiCorrelation_76227079569836
// MI455X (gfx1250) — matrix operand order repaired and checked by reading, not compiled, not run on hardware
//
#include <hip/hip_runtime.h>

typedef __attribute__((ext_vector_type(16))) _Float16 v16h;
typedef __attribute__((ext_vector_type(8)))  _Float16 v8h_t;
typedef __attribute__((ext_vector_type(8)))  float    v8f;
typedef __attribute__((ext_vector_type(4)))  float    v4f;

#define B_TOT 400
#define L_SEQ 96
#define C_DIM 512
#define LC    (L_SEQ * C_DIM)        // 49152
#define N_G1  80
#define N_G0  16
#define JOBS  (B_TOT + N_G0 + N_G1)  // 496
#define P_PAD 104                    // padded row stride (halves): 208B, 16B-aligned

// ---- fragment builders -----------------------------------------------------
// A-matrix 16x32 f16 layout: lane holds row (lane&15); halves 0..7 = K base..base+7,
// halves 8..15 = K base+16..base+23, with base = kchunk + 8*(lane>=16).
__device__ __forceinline__ v16h make_a_from_f32(const float* p) {
  const v4f* q = (const v4f*)p;
  v4f f0 = q[0], f1 = q[1], f2 = q[4], f3 = q[5];
  v16h a;
#pragma unroll
  for (int i = 0; i < 4; ++i) {
    a[i]      = (_Float16)f0[i];
    a[i + 4]  = (_Float16)f1[i];
    a[i + 8]  = (_Float16)f2[i];
    a[i + 12] = (_Float16)f3[i];
  }
  return a;
}

// B-matrix 32x16 f16 layout: lane holds column (lane&15); halves 0..7 = K base..base+7,
// halves 8..15 = K base+16..base+23, with base = kchunk + 8*(lane>=16).
__device__ __forceinline__ v16h make_b_contig(const float* p) {
  const v4f* q = (const v4f*)p;
  v4f f0 = q[0], f1 = q[1], f2 = q[4], f3 = q[5];
  v16h b;
#pragma unroll
  for (int i = 0; i < 4; ++i) {
    b[i]      = (_Float16)f0[i];
    b[i + 4]  = (_Float16)f1[i];
    b[i + 8]  = (_Float16)f2[i];
    b[i + 12] = (_Float16)f3[i];
  }
  return b;
}

// ---- fused attention: one block (12 waves) per 96x512 attention batch ------
__global__ __launch_bounds__(384) void attn_kernel(
    const float* __restrict__ x,     const float* __restrict__ cross,
    const float* __restrict__ xbar0, const float* __restrict__ cbar0,
    const float* __restrict__ xbar1, const float* __restrict__ cbar1,
    float* __restrict__ out, float* __restrict__ o0, float* __restrict__ o1)
{
  __shared__ __align__(16) _Float16 Plds[6][16][P_PAD];   // probabilities (f16)
  __shared__ __align__(16) _Float16 Vt[C_DIM][P_PAD];     // V transposed (f16)

  const int j = blockIdx.x;
  const float* qm; const float* kv; float* om; float oscale;
  if (j < B_TOT) {
    qm = x + (size_t)j * LC;  kv = cross + (size_t)j * LC;
    om = out + (size_t)j * LC;  oscale = 4.0f / 3.0f;   // leaf + level2 contribution
  } else if (j < B_TOT + N_G0) {
    const int g = j - B_TOT;
    qm = xbar0 + (size_t)g * LC;  kv = cbar0 + (size_t)g * LC;
    om = o0 + (size_t)g * LC;  oscale = 1.0f;
  } else {
    const int g = j - B_TOT - N_G0;
    qm = xbar1 + (size_t)g * LC;  kv = cbar1 + (size_t)g * LC;
    om = o1 + (size_t)g * LC;  oscale = 1.0f;
  }

  const int lane = threadIdx.x & 31;
  // force wave index into an SGPR: phase branches become s_cbranch, so EXEC
  // stays all-ones around every WMMA (ISA requirement).
  const int wvu  = __builtin_amdgcn_readfirstlane(threadIdx.x >> 5);  // 0..11
  const int hi   = (lane >> 4) & 1;
  const int l15  = lane & 15;

  if (wvu < 6) {
    // ======== phase 1a (waves 0-5): S = Q*K^T over C=512, then softmax ======
    const int row0 = wvu * 16;
    const v8f vzero = {};
    v8f acc[6];
#pragma unroll
    for (int t = 0; t < 6; ++t) acc[t] = vzero;

    const float* qrp = qm + (size_t)(row0 + l15) * C_DIM;

#pragma unroll 1
    for (int ck = 0; ck < C_DIM; ck += 32) {
      v16h a = make_a_from_f32(qrp + ck + hi * 8);
#pragma unroll
      for (int t = 0; t < 6; ++t) {
        const int n = t * 16 + l15;  // key row index (B column)
        v16h b = make_b_contig(kv + (size_t)n * C_DIM + ck + hi * 8);
        acc[t] = __builtin_amdgcn_wmma_f32_16x16x32_f16(
            false, a, false, b, (short)0, acc[t], false, false);
      }
    }

    // softmax along S=96: each row lives in 16 lanes x 6 tiles of one half-wave
    const float rscale = 0.044194173824159216f;  // 1/sqrt(512)
#pragma unroll
    for (int r = 0; r < 8; ++r) {
      float m = acc[0][r] * rscale;
#pragma unroll
      for (int t = 1; t < 6; ++t) m = fmaxf(m, acc[t][r] * rscale);
#pragma unroll
      for (int msk = 1; msk < 16; msk <<= 1) m = fmaxf(m, __shfl_xor(m, msk, 32));
      float s = 0.0f;
      float e[6];
#pragma unroll
      for (int t = 0; t < 6; ++t) { e[t] = __expf(acc[t][r] * rscale - m); s += e[t]; }
#pragma unroll
      for (int msk = 1; msk < 16; msk <<= 1) s += __shfl_xor(s, msk, 32);
      const float inv = 1.0f / s;
      const int mrow = r + hi * 8;
#pragma unroll
      for (int t = 0; t < 6; ++t)
        Plds[wvu][mrow][t * 16 + l15] = (_Float16)(e[t] * inv);
    }
  } else {
    // ======== phase 1b (waves 6-11): stage V transposed as f16 in LDS =======
    // coalesced float4 reads of V rows; transposed f16 writes (Vt[c][s])
    const int t0 = threadIdx.x - 192;        // 0..191
#pragma unroll 2
    for (int it = 0; it < 64; ++it) {
      const int f  = t0 + it * 192;          // 0..12287 = 96 rows x 128 float4
      const int s  = f >> 7;                 // V row   (0..95)
      const int c0 = (f & 127) << 2;         // channel (0..508, step 4)
      const v4f vv = *(const v4f*)(kv + (size_t)s * C_DIM + c0);
      Vt[c0 + 0][s] = (_Float16)vv[0];
      Vt[c0 + 1][s] = (_Float16)vv[1];
      Vt[c0 + 2][s] = (_Float16)vv[2];
      Vt[c0 + 3][s] = (_Float16)vv[3];
    }
  }

  __syncthreads();

  // ======== phase 2 (all 12 waves): O = P * V from LDS =====================
  const int rt = (wvu >= 6) ? (wvu - 6) : wvu;   // row tile 0..5
  const int ch = (wvu >= 6) ? 1 : 0;             // column half

  // A fragments from Plds (D-layout -> A-layout); two aligned 16B LDS reads each
  v16h pa[3];
#pragma unroll
  for (int scb = 0; scb < 3; ++scb) {
    const int base = scb * 32 + hi * 8;
    const v8h_t h0 = *(const v8h_t*)&Plds[rt][l15][base];
    const v8h_t h1 = *(const v8h_t*)&Plds[rt][l15][base + 16];
    v16h a;
#pragma unroll
    for (int i = 0; i < 8; ++i) { a[i] = h0[i]; a[i + 8] = h1[i]; }
    pa[scb] = a;
  }

  float* orow = om + (size_t)(rt * 16) * C_DIM;
  const v8f vzero2 = {};
#pragma unroll 1
  for (int ct = ch * 16; ct < ch * 16 + 16; ++ct) {
    const int nb = ct * 16 + l15;   // output channel column
    v8f oacc = vzero2;
#pragma unroll
    for (int scb = 0; scb < 3; ++scb) {
      const int s0 = scb * 32 + hi * 8;
      // B fragment: halves S s0..s0+7 and s0+16..s0+23 from transposed V in LDS
      const v8h_t h0 = *(const v8h_t*)&Vt[nb][s0];
      const v8h_t h1 = *(const v8h_t*)&Vt[nb][s0 + 16];
      v16h b;
#pragma unroll
      for (int i = 0; i < 8; ++i) { b[i] = h0[i]; b[i + 8] = h1[i]; }
      oacc = __builtin_amdgcn_wmma_f32_16x16x32_f16(
          false, pa[scb], false, b, (short)0, oacc, false, false);
    }
#pragma unroll
    for (int r = 0; r < 8; ++r)
      orow[(size_t)(r + hi * 8) * C_DIM + nb] = oacc[r] * oscale;
  }
}

// ---- mean over groups of 5 contiguous batches ------------------------------
__global__ void pool5_kernel(const float* __restrict__ src,
                             float* __restrict__ dst, int total) {
  for (int i = blockIdx.x * blockDim.x + threadIdx.x; i < total;
       i += gridDim.x * blockDim.x) {
    const int g = i / LC;
    const int r = i - g * LC;
    const float* s = src + (size_t)g * 5 * LC + r;
    dst[i] = (s[0] + s[LC] + s[2 * LC] + s[3 * LC] + s[4 * LC]) * 0.2f;
  }
}

// ---- out[b] += (o0[b/25] + o1[b/5]) / 3 ------------------------------------
__global__ void combine_kernel(const float* __restrict__ o0,
                               const float* __restrict__ o1,
                               float* __restrict__ out) {
  const int total = B_TOT * LC;
  const float third = 1.0f / 3.0f;
  for (int i = blockIdx.x * blockDim.x + threadIdx.x; i < total;
       i += gridDim.x * blockDim.x) {
    const int b = i / LC;
    const int r = i - b * LC;
    out[i] += (o0[(size_t)(b / 25) * LC + r] + o1[(size_t)(b / 5) * LC + r]) * third;
  }
}

extern "C" void kernel_launch(void* const* d_in, const int* in_sizes, int n_in,
                              void* d_out, int out_size, void* d_ws, size_t ws_size,
                              hipStream_t stream) {
  const float* x     = (const float*)d_in[0];
  const float* cross = (const float*)d_in[1];
  float* out = (float*)d_out;

  float* w = (float*)d_ws;
  float* xbar1 = w;  w += (size_t)N_G1 * LC;
  float* cbar1 = w;  w += (size_t)N_G1 * LC;
  float* xbar0 = w;  w += (size_t)N_G0 * LC;
  float* cbar0 = w;  w += (size_t)N_G0 * LC;
  float* o0    = w;  w += (size_t)N_G0 * LC;
  float* o1    = w;  w += (size_t)N_G1 * LC;

  // hierarchical gather-mean pooling (level1 from raw, level0 from level1)
  pool5_kernel<<<2048, 256, 0, stream>>>(x,     xbar1, N_G1 * LC);
  pool5_kernel<<<2048, 256, 0, stream>>>(cross, cbar1, N_G1 * LC);
  pool5_kernel<<<512,  256, 0, stream>>>(xbar1, xbar0, N_G0 * LC);
  pool5_kernel<<<512,  256, 0, stream>>>(cbar1, cbar0, N_G0 * LC);

  // 496 fused attentions: leaf (scaled 4/3, direct to out) + level0 + level1
  attn_kernel<<<JOBS, 384, 0, stream>>>(x, cross, xbar0, cbar0, xbar1, cbar1,
                                        out, o0, o1);

  // add broadcast hierarchical terms
  combine_kernel<<<8192, 256, 0, stream>>>(o0, o1, out);
}
